// LSTM_13932873908797
// MI455X (gfx1250) — matrix operand order repaired and checked by reading, not compiled, not run on hardware
//
#include <hip/hip_runtime.h>

#define T_STEPS 512
#define BATCH   64
#define IDIM    1024
#define HDIM    1024
#define GDIM    4096   // 4*H
#define KCAT    2048   // I + H

typedef __attribute__((ext_vector_type(16))) _Float16 v16h;
typedef __attribute__((ext_vector_type(8)))  float    v8f;

__device__ __forceinline__ float fast_sigmoid(float x) {
  return 1.0f / (1.0f + __expf(-x));
}
__device__ __forceinline__ float fast_tanh(float x) {
  float e = __expf(2.0f * x);
  return (e - 1.0f) / (e + 1.0f);
}

// ---- CDNA5 async global->LDS copy helpers (ASYNCcnt-tracked) ----
__device__ __forceinline__ void async_g2l_b128(unsigned lds, unsigned long long gaddr) {
  asm volatile("global_load_async_to_lds_b128 %0, %1, off"
               :: "v"(lds), "v"(gaddr) : "memory");
}
__device__ __forceinline__ void async_g2l_b128x2(unsigned lds, unsigned long long gaddr) {
  asm volatile("global_load_async_to_lds_b128 %0, %1, off"
               :: "v"(lds), "v"(gaddr) : "memory");
  asm volatile("global_load_async_to_lds_b128 %0, %1, off offset:16"
               :: "v"(lds), "v"(gaddr) : "memory");
}
__device__ __forceinline__ void wait_async_le(int n) {
  if (n == 0)      asm volatile("s_wait_asynccnt 0x0" ::: "memory");
  else if (n == 1) asm volatile("s_wait_asynccnt 0x1" ::: "memory");
  else             asm volatile("s_wait_asynccnt 0x2" ::: "memory");
}
__device__ __forceinline__ unsigned lds_off(const void* p) {
  return (unsigned)(unsigned long long)p;   // low 32 bits of flat shared addr = LDS offset
}

// ---------------- prep kernels ----------------

__global__ __launch_bounds__(256) void k_conv_inputs(const float* __restrict__ src,
                                                     _Float16* __restrict__ dst, int n) {
  int i = blockIdx.x * 256 + threadIdx.x;
  if (i < n) dst[i] = (_Float16)src[i];
}

// Pack W_cat = [[Wx]; [Wh]] (KCAT x GDIM) into WMMA B-fragments:
// frag(kt, ntg) = 32 lanes x 16 halfs (1KB); lane<16 holds K=kt*32+{0..7,16..23} col ntg*16+lane,
// lane>=16 holds K=kt*32+{8..15,24..31} col ntg*16+(lane-16).
__global__ __launch_bounds__(256) void k_prep_wcat(
    const float* __restrict__ Wxi, const float* __restrict__ Wxf,
    const float* __restrict__ Wxo, const float* __restrict__ Wxc,
    const float* __restrict__ Whi, const float* __restrict__ Whf,
    const float* __restrict__ Who, const float* __restrict__ Whc,
    _Float16* __restrict__ dst) {
  int tid  = blockIdx.x * 256 + threadIdx.x;   // 0 .. KCAT*GDIM-1
  int frag = tid >> 9;                          // 512 halfs / frag
  int idx  = tid & 511;
  int lane = idx >> 4;
  int e    = idx & 15;
  int kt   = frag >> 8;                         // frag = kt*256 + ntg
  int ntg  = frag & 255;
  int kcat = kt * 32 + ((lane < 16) ? 0 : 8) + e + ((e < 8) ? 0 : 8);
  int n    = ntg * 16 + (lane & 15);
  int gate = n >> 10;
  int col  = n & 1023;
  float w;
  if (kcat < IDIM) {
    const float* W = (gate == 0) ? Wxi : (gate == 1) ? Wxf : (gate == 2) ? Wxo : Wxc;
    w = W[kcat * HDIM + col];
  } else {
    const float* W = (gate == 0) ? Whi : (gate == 1) ? Whf : (gate == 2) ? Who : Whc;
    w = W[(kcat - IDIM) * HDIM + col];
  }
  dst[tid] = (_Float16)w;
}

__global__ __launch_bounds__(256) void k_prep_whq(const float* __restrict__ Whq,
                                                  _Float16* __restrict__ dst) {
  int tid  = blockIdx.x * 256 + threadIdx.x;   // 0 .. HDIM*HDIM-1
  int frag = tid >> 9;
  int idx  = tid & 511;
  int lane = idx >> 4;
  int e    = idx & 15;
  int kt   = frag >> 6;                         // frag = kt*64 + ntg
  int ntg  = frag & 63;
  int k    = kt * 32 + ((lane < 16) ? 0 : 8) + e + ((e < 8) ? 0 : 8);
  int n    = ntg * 16 + (lane & 15);
  dst[tid] = (_Float16)Whq[k * HDIM + n];
}

__global__ __launch_bounds__(256) void k_prep_state(
    const float* __restrict__ h0, const float* __restrict__ c0,
    const float* __restrict__ bi, const float* __restrict__ bf,
    const float* __restrict__ bo, const float* __restrict__ bc,
    _Float16* __restrict__ hs0, float* __restrict__ c, float* __restrict__ bcat) {
  int i = blockIdx.x * 256 + threadIdx.x;      // 0 .. BATCH*HDIM-1
  c[i]   = c0[i];
  hs0[i] = (_Float16)h0[i];
  if (i < GDIM) {
    int g = i >> 10, j = i & 1023;
    bcat[i] = (g == 0) ? bi[j] : (g == 1) ? bf[j] : (g == 2) ? bo[j] : bc[j];
  }
}

// ---------------- recurrent step ----------------
// Block = 4 waves sharing col-tile nt (wave w = batch tile mt = staged gate).
// Per k-tile the block async-stages all 4 gate B-fragments (4KB) into LDS once
// (double-buffered one iteration ahead), then every wave reads them from LDS.
__global__ __launch_bounds__(128) void k_lstm_step(
    const _Float16* __restrict__ xt,      // [BATCH, IDIM] f16, this timestep
    const _Float16* __restrict__ hprev,   // [BATCH, HDIM] f16
    _Float16* __restrict__ hout,          // [BATCH, HDIM] f16
    const _Float16* __restrict__ wfrag,   // B-fragments of W_cat
    const float* __restrict__ bcat,       // [GDIM]
    float* __restrict__ c,                // [BATCH, HDIM] fp32 (in/out)
    float* __restrict__ hf32)             // [BATCH, HDIM] fp32 (latest h)
{
  __shared__ _Float16 lbuf[2][4 * 512];   // 2 x 4KB: 4 gate fragments

  const int lane = threadIdx.x & 31;
  const int w    = threadIdx.x >> 5;      // wave id: mt and staged gate
  const int nt   = blockIdx.x;            // 0..63
  const int row  = w * 16 + (lane & 15);
  const int kb   = (lane < 16) ? 0 : 8;

  const _Float16* xrow = xt    + (size_t)row * IDIM;
  const _Float16* hrow = hprev + (size_t)row * HDIM;

  const unsigned long long wbase = (unsigned long long)(const void*)wfrag;
  const unsigned lds_b[2] = { lds_off(&lbuf[0][0]), lds_off(&lbuf[1][0]) };
  const unsigned lds_my   = (unsigned)(w * 1024 + lane * 32);
  // wave w stages gate-w fragment: 32 lanes x 32B = 1KB
  const unsigned long long g_my =
      wbase + (unsigned long long)((w * 64 + nt) * 1024 + lane * 32);

  v8f acc0 = {}, acc1 = {}, acc2 = {}, acc3 = {};

  async_g2l_b128x2(lds_b[0] + lds_my, g_my);            // stage kt=0

  for (int kt = 0; kt < 64; ++kt) {
    // A fragment (global, per-row unique): two contiguous 16B runs
    const _Float16* asrc = (kt < 32) ? (xrow + kt * 32) : (hrow + (kt - 32) * 32);
    const _Float16* ap   = asrc + kb;
    union { v16h v; unsigned u[8]; } a;
#pragma unroll
    for (int i = 0; i < 4; ++i) a.u[i]     = *(const unsigned*)(ap + 2 * i);
#pragma unroll
    for (int i = 0; i < 4; ++i) a.u[4 + i] = *(const unsigned*)(ap + 16 + 2 * i);

    if (kt + 1 < 64) {                                   // stage kt+1, keep kt in flight
      async_g2l_b128x2(lds_b[(kt + 1) & 1] + lds_my,
                       g_my + (unsigned long long)(kt + 1) * (256 * 1024));
      wait_async_le(2);                                  // kt's two copies done
    } else {
      wait_async_le(0);
    }
    __syncthreads();                                     // all 4 gates visible

    const _Float16* bsrc = &lbuf[kt & 1][0] + lane * 16;
    v16h b0 = *(const v16h*)(bsrc + 0 * 512);
    v16h b1 = *(const v16h*)(bsrc + 1 * 512);
    v16h b2 = *(const v16h*)(bsrc + 2 * 512);
    v16h b3 = *(const v16h*)(bsrc + 3 * 512);

    acc0 = __builtin_amdgcn_wmma_f32_16x16x32_f16(false, a.v, false, b0, (short)0, acc0, false, false);
    acc1 = __builtin_amdgcn_wmma_f32_16x16x32_f16(false, a.v, false, b1, (short)0, acc1, false, false);
    acc2 = __builtin_amdgcn_wmma_f32_16x16x32_f16(false, a.v, false, b2, (short)0, acc2, false, false);
    acc3 = __builtin_amdgcn_wmma_f32_16x16x32_f16(false, a.v, false, b3, (short)0, acc3, false, false);

    __syncthreads();                                     // safe to overwrite other buffer
  }

  // C/D layout: lane holds col n = nt*16 + (lane&15); vgpr r holds row
  // m = mt*16 + r + (lane<16 ? 0 : 8).
  const int n     = nt * 16 + (lane & 15);
  const int mrow0 = w * 16 + ((lane >> 4) << 3);
  const float bi = bcat[0 * HDIM + n];
  const float bf = bcat[1 * HDIM + n];
  const float bo = bcat[2 * HDIM + n];
  const float bc = bcat[3 * HDIM + n];

#pragma unroll
  for (int r = 0; r < 8; ++r) {
    const int m = mrow0 + r;
    const float gi = acc0[r] + bi;
    const float gf = acc1[r] + bf;
    const float go = acc2[r] + bo;
    const float gc = acc3[r] + bc;
    const float si = fast_sigmoid(gi);
    const float sf = fast_sigmoid(gf);
    const float so = fast_sigmoid(go);
    const float ct = fast_tanh(gc);
    const float cn = sf * c[m * HDIM + n] + si * ct;
    const float hn = so * fast_tanh(cn);
    c[m * HDIM + n]    = cn;
    hf32[m * HDIM + n] = hn;
    hout[m * HDIM + n] = (_Float16)hn;
  }
}

// ---------------- output GEMM ----------------
// Block = 8 waves (row tiles) sharing one quad of N-tiles; per k-tile the block
// async-stages 4 B-fragments into LDS (each wave copies 512B); each wave holds
// 4 accumulators (A reuse x4).
__global__ __launch_bounds__(256) void k_out_gemm(
    const _Float16* __restrict__ hs,    // [(T+1)*BATCH, HDIM]; rows 64.. are h_1..h_T
    const _Float16* __restrict__ whq,   // B-fragments
    const float* __restrict__ bq,       // [HDIM]
    float* __restrict__ out)            // [T*BATCH, HDIM]
{
  __shared__ _Float16 lbuf[2][4 * 512];   // 2 x 4KB: 4 N-tile fragments

  const int lane = threadIdx.x & 31;
  const int w    = threadIdx.x >> 5;      // 0..7: row tile within block
  const int ntq  = blockIdx.x & 15;       // quad of ntg: ntg = ntq*4+g
  const int rtg  = blockIdx.x >> 4;       // 0..255
  const int rt   = rtg * 8 + w;           // 0..2047
  const int rowA = BATCH + rt * 16 + (lane & 15);
  const int kb   = (lane < 16) ? 0 : 8;
  const _Float16* arow = hs + (size_t)rowA * HDIM;

  const unsigned long long wbase = (unsigned long long)(const void*)whq;
  const unsigned lds_b[2] = { lds_off(&lbuf[0][0]), lds_off(&lbuf[1][0]) };
  // wave w copies bytes [w*512, w*512+512) of the 4KB block: frag g=w>>1, half w&1
  const unsigned lds_my = (unsigned)(w * 512 + lane * 16);
  const unsigned long long g_my =
      wbase + (unsigned long long)((ntq * 4 + (w >> 1)) * 1024 + (w & 1) * 512 + lane * 16);

  v8f acc0 = {}, acc1 = {}, acc2 = {}, acc3 = {};

  async_g2l_b128(lds_b[0] + lds_my, g_my);               // stage kt=0

  for (int kt = 0; kt < 32; ++kt) {
    const _Float16* ap = arow + kt * 32 + kb;
    union { v16h v; unsigned u[8]; } a;
#pragma unroll
    for (int i = 0; i < 4; ++i) a.u[i]     = *(const unsigned*)(ap + 2 * i);
#pragma unroll
    for (int i = 0; i < 4; ++i) a.u[4 + i] = *(const unsigned*)(ap + 16 + 2 * i);
    __builtin_prefetch(ap + 32, 0, 1);

    if (kt + 1 < 32) {
      async_g2l_b128(lds_b[(kt + 1) & 1] + lds_my,
                     g_my + (unsigned long long)(kt + 1) * (64 * 1024));
      wait_async_le(1);
    } else {
      wait_async_le(0);
    }
    __syncthreads();

    const _Float16* bsrc = &lbuf[kt & 1][0] + lane * 16;
    v16h b0 = *(const v16h*)(bsrc + 0 * 512);
    v16h b1 = *(const v16h*)(bsrc + 1 * 512);
    v16h b2 = *(const v16h*)(bsrc + 2 * 512);
    v16h b3 = *(const v16h*)(bsrc + 3 * 512);

    acc0 = __builtin_amdgcn_wmma_f32_16x16x32_f16(false, a.v, false, b0, (short)0, acc0, false, false);
    acc1 = __builtin_amdgcn_wmma_f32_16x16x32_f16(false, a.v, false, b1, (short)0, acc1, false, false);
    acc2 = __builtin_amdgcn_wmma_f32_16x16x32_f16(false, a.v, false, b2, (short)0, acc2, false, false);
    acc3 = __builtin_amdgcn_wmma_f32_16x16x32_f16(false, a.v, false, b3, (short)0, acc3, false, false);

    __syncthreads();
  }

  const int orow0 = rt * 16 + ((lane >> 4) << 3);
#pragma unroll
  for (int g = 0; g < 4; ++g) {
    const int n   = (ntq * 4 + g) * 16 + (lane & 15);
    const float bb = bq[n];
    const v8f acc = (g == 0) ? acc0 : (g == 1) ? acc1 : (g == 2) ? acc2 : acc3;
#pragma unroll
    for (int r = 0; r < 8; ++r)
      out[(size_t)(orow0 + r) * HDIM + n] = acc[r] + bb;
  }
}

__global__ __launch_bounds__(256) void k_finalize(const float* __restrict__ hf,
                                                  const float* __restrict__ c,
                                                  float* __restrict__ out) {
  int i = blockIdx.x * 256 + threadIdx.x;     // 0 .. BATCH*HDIM-1
  const size_t base = (size_t)T_STEPS * BATCH * HDIM;
  out[base + i]                = hf[i];
  out[base + BATCH * HDIM + i] = c[i];
}

// ---------------- launch ----------------

extern "C" void kernel_launch(void* const* d_in, const int* in_sizes, int n_in,
                              void* d_out, int out_size, void* d_ws, size_t ws_size,
                              hipStream_t stream) {
  (void)in_sizes; (void)n_in; (void)out_size; (void)ws_size;
  const float* inputs = (const float*)d_in[0];
  const float* h0     = (const float*)d_in[1];
  const float* c0     = (const float*)d_in[2];
  const float* W_xi   = (const float*)d_in[3];
  const float* W_hi   = (const float*)d_in[4];
  const float* b_i    = (const float*)d_in[5];
  const float* W_xf   = (const float*)d_in[6];
  const float* W_hf   = (const float*)d_in[7];
  const float* b_f    = (const float*)d_in[8];
  const float* W_xo   = (const float*)d_in[9];
  const float* W_ho   = (const float*)d_in[10];
  const float* b_o    = (const float*)d_in[11];
  const float* W_xc   = (const float*)d_in[12];
  const float* W_hc   = (const float*)d_in[13];
  const float* b_c    = (const float*)d_in[14];
  const float* W_hq   = (const float*)d_in[15];
  const float* b_q    = (const float*)d_in[16];
  float* out = (float*)d_out;

  char* ws = (char*)d_ws;
  size_t off = 0;
  _Float16* x16  = (_Float16*)(ws + off); off += (size_t)T_STEPS * BATCH * IDIM * 2;
  _Float16* wcat = (_Float16*)(ws + off); off += (size_t)KCAT * GDIM * 2;
  _Float16* whq  = (_Float16*)(ws + off); off += (size_t)HDIM * HDIM * 2;
  float*    bcat = (float*)(ws + off);    off += (size_t)GDIM * 4;
  _Float16* hs   = (_Float16*)(ws + off); off += (size_t)(T_STEPS + 1) * BATCH * HDIM * 2;
  float*    cbuf = (float*)(ws + off);    off += (size_t)BATCH * HDIM * 4;
  float*    hf32 = (float*)(ws + off);    off += (size_t)BATCH * HDIM * 4;

  k_conv_inputs<<<(T_STEPS * BATCH * IDIM) / 256, 256, 0, stream>>>(
      inputs, x16, T_STEPS * BATCH * IDIM);
  k_prep_wcat<<<(KCAT * GDIM) / 256, 256, 0, stream>>>(
      W_xi, W_xf, W_xo, W_xc, W_hi, W_hf, W_ho, W_hc, wcat);
  k_prep_whq<<<(HDIM * HDIM) / 256, 256, 0, stream>>>(W_hq, whq);
  k_prep_state<<<(BATCH * HDIM) / 256, 256, 0, stream>>>(
      h0, c0, b_i, b_f, b_o, b_c, hs, cbuf, bcat);

  for (int t = 0; t < T_STEPS; ++t) {
    k_lstm_step<<<64, 128, 0, stream>>>(
        x16 + (size_t)t * BATCH * IDIM,
        hs + (size_t)t * BATCH * HDIM,
        hs + (size_t)(t + 1) * BATCH * HDIM,
        wcat, bcat, cbuf, hf32);
  }

  k_out_gemm<<<16 * 256, 256, 0, stream>>>(hs, whq, b_q, out);
  k_finalize<<<(BATCH * HDIM) / 256, 256, 0, stream>>>(hf32, cbuf, out);
}
